// SparseDiffMlp_66752381714947
// MI455X (gfx1250) — matrix operand order repaired and checked by reading, not compiled, not run on hardware
//
#include <hip/hip_runtime.h>

// ---------- CDNA5 vector types ----------
typedef __attribute__((ext_vector_type(16))) __bf16 v16bf;
typedef __attribute__((ext_vector_type(8)))  __bf16 v8bf;
typedef __attribute__((ext_vector_type(4)))  __bf16 v4bf;
typedef __attribute__((ext_vector_type(8)))  float  v8f;

#define N_TOK 4096
#define C_DIM 3072
#define F_DIM 12288
#define MBM   16
#define BM    128
#define NBLK  32          // N_TOK / BM
#define MBS   256         // N_TOK / MBM
#define KSEL  1536        // F_DIM / 8
#define LDSK  40          // padded LDS row stride (bf16 elems) = 80 bytes, 16B aligned

// ---------- helpers ----------
__device__ __forceinline__ __bf16 to_bf16(float f) {
  unsigned u = __float_as_uint(f);
  unsigned r = u + 0x7FFFu + ((u >> 16) & 1u);   // round-to-nearest-even
  unsigned short h = (unsigned short)(r >> 16);
  __bf16 b;
  __builtin_memcpy(&b, &h, sizeof(h));
  return b;
}

// A operand (16x32 bf16): lanes 0-15 hold K {0..7,16..23}; lanes 16-31 hold K {8..15,24..31}
__device__ __forceinline__ v16bf fetchA(const __bf16* row, int lane) {
  int bk = (lane & 16) ? 8 : 0;
  v8bf lo = *(const v8bf*)(row + bk);
  v8bf hi = *(const v8bf*)(row + 16 + bk);
  return __builtin_shufflevector(lo, hi, 0,1,2,3,4,5,6,7,8,9,10,11,12,13,14,15);
}

// B operand (32x16 bf16, stored as [N][K] rows): lanes 0-15 hold K {0..7,16..23}; lanes 16-31 hold K {8..15,24..31}
__device__ __forceinline__ v16bf fetchB(const __bf16* row, int lane) {
  int off = (lane & 16) ? 8 : 0;
  v8bf lo = *(const v8bf*)(row + off);
  v8bf hi = *(const v8bf*)(row + off + 16);
  return __builtin_shufflevector(lo, hi, 0,1,2,3,4,5,6,7,8,9,10,11,12,13,14,15);
}

__device__ __forceinline__ v8f wmma_bf16(v16bf a, v16bf b, v8f c) {
  return __builtin_amdgcn_wmma_f32_16x16x32_bf16(false, a, false, b, (short)0, c, false, false);
}

__device__ __forceinline__ float gelu_tanh(float x) {
  float x3 = x * x * x;
  return 0.5f * x * (1.0f + tanhf(0.7978845608028654f * (x + 0.044715f * x3)));
}

// ---------- kernel 0: zero mdiff ----------
__global__ void sdm_zero_kernel(float* p, int n) {
  int i = blockIdx.x * blockDim.x + threadIdx.x;
  if (i < n) p[i] = 0.0f;
}

// ---------- kernel 1: block-mean of x ----------
__global__ __launch_bounds__(256)
void sdm_blockmean_kernel(const float* __restrict__ x, float* __restrict__ bm) {
  int i  = blockIdx.x * 256 + threadIdx.x;     // over MBS*C_DIM
  int mb = i / C_DIM;
  int c  = i - mb * C_DIM;
  const float* p = x + (size_t)mb * MBM * C_DIM + c;
  float s = 0.0f;
#pragma unroll
  for (int t = 0; t < MBM; ++t) s += p[(size_t)t * C_DIM];
  bm[i] = s * (1.0f / (float)MBM);
}

// ---------- kernel 2: bmfc1 GEMM (256x12288, K=3072) fused abs-diff -> mdiff ----------
__global__ __launch_bounds__(256)
void sdm_gemm_mdiff(const float* __restrict__ bmx, const float* __restrict__ fc1w,
                    const float* __restrict__ fc1b, const float* __restrict__ bmc,
                    float* __restrict__ mdiff) {
  __shared__ __bf16 As[BM * LDSK];
  __shared__ __bf16 Bs[BM * LDSK];
  const int n0 = blockIdx.x * 128;
  const int m0 = blockIdx.y * 128;
  const int tid  = threadIdx.x;
  const int lane = tid & 31;
  const int w  = tid >> 5;
  const int wm = w & 1;
  const int wn = w >> 1;
  const int lr = tid >> 3;
  const int cs = (tid & 7) * 4;

  v8f acc[4][2];
  v8f vz = {0.f,0.f,0.f,0.f,0.f,0.f,0.f,0.f};
#pragma unroll
  for (int s = 0; s < 4; ++s)
#pragma unroll
    for (int n = 0; n < 2; ++n) acc[s][n] = vz;

  for (int k0 = 0; k0 < C_DIM; k0 += 32) {
    __syncthreads();
#pragma unroll
    for (int i = 0; i < 4; ++i) {
      int row = lr + i * 32;
      float4 va = *(const float4*)(bmx + (size_t)(m0 + row) * C_DIM + k0 + cs);
      v4bf a4 = { to_bf16(va.x), to_bf16(va.y), to_bf16(va.z), to_bf16(va.w) };
      *(v4bf*)(As + row * LDSK + cs) = a4;
      float4 vb = *(const float4*)(fc1w + (size_t)(n0 + row) * C_DIM + k0 + cs);
      v4bf b4 = { to_bf16(vb.x), to_bf16(vb.y), to_bf16(vb.z), to_bf16(vb.w) };
      *(v4bf*)(Bs + row * LDSK + cs) = b4;
    }
    __syncthreads();
    v16bf av[4], bv[2];
#pragma unroll
    for (int s = 0; s < 4; ++s)
      av[s] = fetchA(As + (wm * 64 + s * 16 + (lane & 15)) * LDSK, lane);
#pragma unroll
    for (int n = 0; n < 2; ++n)
      bv[n] = fetchB(Bs + (wn * 32 + n * 16 + (lane & 15)) * LDSK, lane);
#pragma unroll
    for (int s = 0; s < 4; ++s)
#pragma unroll
      for (int n = 0; n < 2; ++n)
        acc[s][n] = wmma_bf16(av[s], bv[n], acc[s][n]);
  }

#pragma unroll
  for (int s = 0; s < 4; ++s) {
    int rowb = m0 + wm * 64 + s * 16 + ((lane & 16) ? 8 : 0);   // mean-block row, mult of 8
#pragma unroll
    for (int n = 0; n < 2; ++n) {
      int fcol = n0 + wn * 32 + n * 16 + (lane & 15);
      float bias = fc1b[fcol];
      const float* cp = bmc + (size_t)rowb * F_DIM + fcol;
      float ssum = 0.0f;
#pragma unroll
      for (int e = 0; e < 8; ++e)
        ssum += fabsf(acc[s][n][e] + bias - cp[(size_t)e * F_DIM]);
      atomicAdd(mdiff + (size_t)(rowb >> 3) * F_DIM + fcol, ssum);
    }
  }
}

// ---------- kernel 3: per-block top-k radix select ----------
__global__ __launch_bounds__(256)
void sdm_topk_kernel(const float* __restrict__ mdiff, int* __restrict__ inds) {
  __shared__ int hist[4096];
  __shared__ int s_thresh, s_gt, s_c0, s_c1;
  const int m = blockIdx.x;
  const int tid = threadIdx.x;
  const float* v = mdiff + (size_t)m * F_DIM;
  for (int i = tid; i < 4096; i += 256) hist[i] = 0;
  __syncthreads();
  for (int i = tid; i < F_DIM; i += 256) {
    unsigned b = __float_as_uint(v[i]) >> 20;   // values >= 0 -> monotonic bits
    atomicAdd(&hist[b], 1);
  }
  __syncthreads();
  if (tid == 0) {
    int cum = 0, t = 0;
    for (int b = 4095; b >= 0; --b) {
      int c = hist[b];
      if (cum + c >= KSEL) { t = b; break; }
      cum += c;
    }
    s_thresh = t; s_gt = cum; s_c0 = 0; s_c1 = 0;
  }
  __syncthreads();
  const int T = s_thresh, gt = s_gt, quota = KSEL - s_gt;
  for (int i = tid; i < F_DIM; i += 256) {
    int b = (int)(__float_as_uint(v[i]) >> 20);
    if (b > T) {
      int p = atomicAdd(&s_c0, 1);
      inds[m * KSEL + p] = i;
    } else if (b == T) {
      int p = atomicAdd(&s_c1, 1);
      if (p < quota) inds[m * KSEL + gt + p] = i;
    }
  }
}

// ---------- kernel 4: sparse fc1 GEMM + GELU + delta-vs-cache -> bf16 delta ----------
__global__ __launch_bounds__(256)
void sdm_gemm_delta(const float* __restrict__ x, const float* __restrict__ fc1w,
                    const float* __restrict__ fc1b, const int* __restrict__ inds,
                    const float* __restrict__ sactT, __bf16* __restrict__ delta) {
  __shared__ __bf16 As[BM * LDSK];
  __shared__ __bf16 Bs[BM * LDSK];
  const int m  = blockIdx.y;
  const int n0 = blockIdx.x * 128;
  const int tid  = threadIdx.x;
  const int lane = tid & 31;
  const int w  = tid >> 5;
  const int wm = w & 1;
  const int wn = w >> 1;
  const int lr = tid >> 3;
  const int cs = (tid & 7) * 4;

  const float* Asrc = x + (size_t)m * BM * C_DIM;
  const int* idxp = inds + m * KSEL + n0;

  v8f acc[4][2];
  v8f vz = {0.f,0.f,0.f,0.f,0.f,0.f,0.f,0.f};
#pragma unroll
  for (int s = 0; s < 4; ++s)
#pragma unroll
    for (int n = 0; n < 2; ++n) acc[s][n] = vz;

  for (int k0 = 0; k0 < C_DIM; k0 += 32) {
    __syncthreads();
#pragma unroll
    for (int i = 0; i < 4; ++i) {
      int row = lr + i * 32;
      float4 va = *(const float4*)(Asrc + (size_t)row * C_DIM + k0 + cs);
      v4bf a4 = { to_bf16(va.x), to_bf16(va.y), to_bf16(va.z), to_bf16(va.w) };
      *(v4bf*)(As + row * LDSK + cs) = a4;
      int gidx = idxp[row];
      float4 vb = *(const float4*)(fc1w + (size_t)gidx * C_DIM + k0 + cs);
      v4bf b4 = { to_bf16(vb.x), to_bf16(vb.y), to_bf16(vb.z), to_bf16(vb.w) };
      *(v4bf*)(Bs + row * LDSK + cs) = b4;
    }
    __syncthreads();
    v16bf av[4], bv[2];
#pragma unroll
    for (int s = 0; s < 4; ++s)
      av[s] = fetchA(As + (wm * 64 + s * 16 + (lane & 15)) * LDSK, lane);
#pragma unroll
    for (int n = 0; n < 2; ++n)
      bv[n] = fetchB(Bs + (wn * 32 + n * 16 + (lane & 15)) * LDSK, lane);
#pragma unroll
    for (int s = 0; s < 4; ++s)
#pragma unroll
      for (int n = 0; n < 2; ++n)
        acc[s][n] = wmma_bf16(av[s], bv[n], acc[s][n]);
  }

#pragma unroll
  for (int s = 0; s < 4; ++s) {
    int trow = wm * 64 + s * 16 + ((lane & 16) ? 8 : 0);
#pragma unroll
    for (int n = 0; n < 2; ++n) {
      int jcol = n0 + wn * 32 + n * 16 + (lane & 15);
      int gidx = inds[m * KSEL + jcol];
      float bias = fc1b[gidx];
      const float* sp = sactT + (size_t)gidx * N_TOK + m * BM + trow;
      __bf16* dp = delta + (size_t)(m * BM + trow) * KSEL + jcol;
#pragma unroll
      for (int e = 0; e < 8; ++e) {
        float mid = acc[s][n][e] + bias;
        float d = gelu_tanh(mid) - sp[e];
        dp[(size_t)e * KSEL] = to_bf16(d);
      }
    }
  }
}

// ---------- kernel 5: sparse fc2 GEMM + out_cache add ----------
__global__ __launch_bounds__(256)
void sdm_gemm_out(const __bf16* __restrict__ delta, const float* __restrict__ fc2wT,
                  const int* __restrict__ inds, const float* __restrict__ outc,
                  float* __restrict__ out) {
  __shared__ __bf16 As[BM * LDSK];
  __shared__ __bf16 Bs[BM * LDSK];
  const int m  = blockIdx.y;
  const int c0 = blockIdx.x * 128;
  const int tid  = threadIdx.x;
  const int lane = tid & 31;
  const int w  = tid >> 5;
  const int wm = w & 1;
  const int wn = w >> 1;
  const int lr = tid >> 3;
  const int cs = (tid & 7) * 4;
  const int jj = tid >> 3;          // 0..31 over K rows of fc2 tile
  const int c8 = tid & 7;

  const __bf16* Asrc = delta + (size_t)m * BM * KSEL;

  v8f acc[4][2];
  v8f vz = {0.f,0.f,0.f,0.f,0.f,0.f,0.f,0.f};
#pragma unroll
  for (int s = 0; s < 4; ++s)
#pragma unroll
    for (int n = 0; n < 2; ++n) acc[s][n] = vz;

  for (int j0 = 0; j0 < KSEL; j0 += 32) {
    __syncthreads();
    // A: 128(t) x 32(j) bf16, already bf16 in ws
#pragma unroll
    for (int i = 0; i < 4; ++i) {
      int row = lr + i * 32;
      v4bf a4 = *(const v4bf*)(Asrc + (size_t)row * KSEL + j0 + cs);
      *(v4bf*)(As + row * LDSK + cs) = a4;
    }
    // B: gather fc2w_T rows (K=j) and transpose into [N=c][K=j] layout
    {
      int gidx = inds[m * KSEL + j0 + jj];
      const float* wrow = fc2wT + (size_t)gidx * C_DIM + c0;
#pragma unroll
      for (int i = 0; i < 4; ++i) {
        int c = c8 * 4 + i * 32;
        float4 vb = *(const float4*)(wrow + c);
        Bs[(c + 0) * LDSK + jj] = to_bf16(vb.x);
        Bs[(c + 1) * LDSK + jj] = to_bf16(vb.y);
        Bs[(c + 2) * LDSK + jj] = to_bf16(vb.z);
        Bs[(c + 3) * LDSK + jj] = to_bf16(vb.w);
      }
    }
    __syncthreads();
    v16bf av[4], bv[2];
#pragma unroll
    for (int s = 0; s < 4; ++s)
      av[s] = fetchA(As + (wm * 64 + s * 16 + (lane & 15)) * LDSK, lane);
#pragma unroll
    for (int n = 0; n < 2; ++n)
      bv[n] = fetchB(Bs + (wn * 32 + n * 16 + (lane & 15)) * LDSK, lane);
#pragma unroll
    for (int s = 0; s < 4; ++s)
#pragma unroll
      for (int n = 0; n < 2; ++n)
        acc[s][n] = wmma_bf16(av[s], bv[n], acc[s][n]);
  }

#pragma unroll
  for (int s = 0; s < 4; ++s) {
    int trow = wm * 64 + s * 16 + ((lane & 16) ? 8 : 0);
#pragma unroll
    for (int n = 0; n < 2; ++n) {
      int ccol = c0 + wn * 32 + n * 16 + (lane & 15);
      const float* op = outc + (size_t)(m * BM + trow) * C_DIM + ccol;
      float* dp = out + (size_t)(m * BM + trow) * C_DIM + ccol;
#pragma unroll
      for (int e = 0; e < 8; ++e)
        dp[(size_t)e * C_DIM] = op[(size_t)e * C_DIM] + acc[s][n][e];
    }
  }
}

// ---------- launch ----------
extern "C" void kernel_launch(void* const* d_in, const int* in_sizes, int n_in,
                              void* d_out, int out_size, void* d_ws, size_t ws_size,
                              hipStream_t stream) {
  const float* x     = (const float*)d_in[0];
  const float* fc1w  = (const float*)d_in[1];
  const float* fc1b  = (const float*)d_in[2];
  const float* fc2wT = (const float*)d_in[3];
  const float* bmc   = (const float*)d_in[4];
  const float* sactT = (const float*)d_in[5];
  const float* outc  = (const float*)d_in[6];
  float* out = (float*)d_out;

  // workspace layout (~16.7 MB)
  float*  bmx   = (float*)d_ws;                          // 256*3072 f32
  float*  mdiff = bmx + (size_t)MBS * C_DIM;             // 32*12288 f32
  int*    inds  = (int*)(mdiff + (size_t)NBLK * F_DIM);  // 32*1536 i32
  __bf16* delta = (__bf16*)(inds + (size_t)NBLK * KSEL); // 32*128*1536 bf16

  sdm_zero_kernel<<<(NBLK * F_DIM + 255) / 256, 256, 0, stream>>>(mdiff, NBLK * F_DIM);
  sdm_blockmean_kernel<<<(MBS * C_DIM) / 256, 256, 0, stream>>>(x, bmx);
  sdm_gemm_mdiff<<<dim3(F_DIM / 128, MBS / 128), 256, 0, stream>>>(bmx, fc1w, fc1b, bmc, mdiff);
  sdm_topk_kernel<<<NBLK, 256, 0, stream>>>(mdiff, inds);
  sdm_gemm_delta<<<dim3(KSEL / 128, NBLK), 256, 0, stream>>>(x, fc1w, fc1b, inds, sactT, delta);
  sdm_gemm_out<<<dim3(C_DIM / 128, NBLK), 256, 0, stream>>>(delta, fc2wT, inds, outc, out);
}
